// SimpleRNN_39582418600359
// MI455X (gfx1250) — matrix operand order repaired and checked by reading, not compiled, not run on hardware
//
#include <hip/hip_runtime.h>

#define BB   128      // batch
#define SS   1024     // seq
#define HH   2048     // hidden
#define VV   256      // vocab
#define NB   64       // persistent blocks: 8 mtiles x 8 nslabs
#define TPB  256      // 8 waves of 32
#define NWAVES (NB * (TPB / 32))

#define LDS_ROW_BYTES (HH * 2 + 16)    // 4 KB row + 16B pad to spread LDS banks

typedef __attribute__((ext_vector_type(16))) __bf16 v16bf;
typedef __attribute__((ext_vector_type(8)))  __bf16 v8bf;
typedef __attribute__((ext_vector_type(8)))  float  v8f;

// ---------------- preprocessing: pack weights into WMMA B-fragment layout ----
// B-fragment (16-bit, 32x16): lane L<16 holds B[K=e<8 ? e : e+8][N=L], lanes 16-31 K+8.
// Packed so each lane loads its 16 bf16 contiguously (2 x b128).
__global__ void pack_whh(const float* __restrict__ w, __bf16* __restrict__ o) {
    int idx = blockIdx.x * blockDim.x + threadIdx.x;     // H*H elements
    if (idx >= HH * HH) return;
    int e     = idx & 15;
    int lane  = (idx >> 4) & 31;
    int kc    = (idx >> 9) & 63;
    int ntile = idx >> 15;                               // 0..127
    int k = kc * 32 + ((e & 8) << 1) + ((lane & 16) >> 1) + (e & 7);
    int n = ntile * 16 + (lane & 15);
    o[idx] = (__bf16)w[k * HH + n];
}

__global__ void pack_wlin(const float* __restrict__ w, __bf16* __restrict__ o) {
    int idx = blockIdx.x * blockDim.x + threadIdx.x;     // H*V elements
    if (idx >= HH * VV) return;
    int e     = idx & 15;
    int lane  = (idx >> 4) & 31;
    int kc    = (idx >> 9) & 63;
    int ntile = idx >> 15;                               // 0..15
    int k = kc * 32 + ((e & 8) << 1) + ((lane & 16) >> 1) + (e & 7);
    int n = ntile * 16 + (lane & 15);
    o[idx] = (__bf16)w[k * VV + n];
}

__global__ void cvt_h0(const float* __restrict__ hin, __bf16* __restrict__ hout) {
    int i = blockIdx.x * blockDim.x + threadIdx.x;
    if (i < BB * HH) hout[i] = (__bf16)hin[i];
}

// ---------------- helpers ----------------------------------------------------
__device__ __forceinline__ float fast_tanh(float x) {
    // branch-free: tanh(x) = sign(x) * (1 - e) / (1 + e),  e = exp(-2|x|)
    float ax = __builtin_fabsf(x);
    float e  = __expf(-2.0f * ax);                    // v_exp_f32 pipe
    float r  = (1.0f - e) * __builtin_amdgcn_rcpf(1.0f + e);
    return __builtin_copysignf(r, x);
}

__device__ __forceinline__ void grid_barrier(unsigned* cnt, unsigned target) {
    __threadfence();
    __syncthreads();
    if (threadIdx.x == 0) {
        __hip_atomic_fetch_add(cnt, 1u, __ATOMIC_RELEASE, __HIP_MEMORY_SCOPE_AGENT);
        while (__hip_atomic_load(cnt, __ATOMIC_ACQUIRE, __HIP_MEMORY_SCOPE_AGENT) < target)
            __builtin_amdgcn_s_sleep(1);
    }
    __syncthreads();
}

// ---------------- persistent RNN kernel -------------------------------------
__global__ __launch_bounds__(TPB, 1)
void rnn_persistent(const int*   __restrict__ x,       // (B,S) int32
                    const float* __restrict__ W_xh,    // (V,H) fp32
                    const float* __restrict__ b_h,     // (H,)
                    const float* __restrict__ b_lin,   // (V,)
                    const __bf16* __restrict__ Whh_p,  // packed (128 ntile)(64 kc)(32 lane)(16)
                    const __bf16* __restrict__ Wlin_p, // packed (16 ntile)(64 kc)(32 lane)(16)
                    __bf16* __restrict__ hbuf,         // 2 * B*H bf16, double-buffered
                    float*  __restrict__ out,          // (B,S,V)
                    float*  __restrict__ hlast,        // (B,H)
                    unsigned* __restrict__ bar) {
    __shared__ __align__(16) unsigned char smemA[16 * LDS_ROW_BYTES];

    const int lane  = threadIdx.x & 31;
    const int wave  = threadIdx.x >> 5;
    const int gwave = blockIdx.x * (TPB / 32) + wave;
    const int half  = lane >> 4;
    const int l15   = lane & 15;
    const int mtile = blockIdx.x >> 3;   // 0..7   (16 batch rows)
    const int nslab = blockIdx.x & 7;    // 0..7   (256 hidden cols)

    const unsigned smem_base = (unsigned)(size_t)(void*)smemA;  // LDS byte offset

    for (int t = 0; t < SS; ++t) {
        const __bf16* hc = hbuf + (t & 1) * (BB * HH);
        __bf16*       hn = hbuf + ((t + 1) & 1) * (BB * HH);

        // -------- stage h(t) rows [mtile*16, +16) into LDS via async DMA -----
        {
            const unsigned char* gbase = (const unsigned char*)(hc + mtile * 16 * HH);
            for (int c = threadIdx.x; c < 16 * 256; c += TPB) {   // 16B chunks
                int row = c >> 8, col = c & 255;
                unsigned     loff = smem_base + row * LDS_ROW_BYTES + col * 16;
                const unsigned char* g = gbase + row * (HH * 2) + col * 16;
                asm volatile("global_load_async_to_lds_b128 %0, %1, off"
                             :: "v"(loff), "v"(g) : "memory");
            }
            asm volatile("s_wait_asynccnt 0x0" ::: "memory");
            __syncthreads();
        }

        // -------- phase 1: h_new = tanh(emb + h @ W_hh + b_h) ----------------
        // each wave: one A-fragment feeds 2 n-tiles (2 WMMAs per kc)
        {
            const int n0 = nslab * 16 + wave * 2;      // global ntile 0..127
            v8f acc0 = {}, acc1 = {};
            const unsigned char* arow = smemA + l15 * LDS_ROW_BYTES + half * 16;
            const __bf16* bp0 = Whh_p + (n0 * 64) * 512 + lane * 16;
            const __bf16* bp1 = bp0 + 64 * 512;        // next ntile
            for (int kc = 0; kc < HH / 32; ++kc) {
                v8bf alo = *(const v8bf*)(arow + kc * 64);        // K +0..7
                v8bf ahi = *(const v8bf*)(arow + kc * 64 + 32);   // K +16..23
                v16bf a;
#pragma unroll
                for (int i = 0; i < 8; ++i) { a[i] = alo[i]; a[8 + i] = ahi[i]; }
                v16bf b0 = *(const v16bf*)(bp0 + kc * 512);
                v16bf b1 = *(const v16bf*)(bp1 + kc * 512);
                acc0 = __builtin_amdgcn_wmma_f32_16x16x32_bf16(
                           false, a, false, b0, (short)0, acc0, false, false);
                acc1 = __builtin_amdgcn_wmma_f32_16x16x32_bf16(
                           false, a, false, b1, (short)0, acc1, false, false);
            }
#pragma unroll
            for (int nt = 0; nt < 2; ++nt) {
                v8f acc = nt ? acc1 : acc0;
                const int col = (n0 + nt) * 16 + l15;
                const float bh = b_h[col];
#pragma unroll
                for (int r = 0; r < 8; ++r) {
                    const int row = mtile * 16 + half * 8 + r;   // C/D: M = r (+8 hi half)
                    const int tok = x[row * SS + t];
                    const float v = fast_tanh(acc[r] + W_xh[tok * HH + col] + bh);
                    hn[row * HH + col] = (__bf16)v;
                    if (t == SS - 1) hlast[row * HH + col] = v;
                }
            }
        }

        // one grid-wide barrier per timestep (double buffer makes it sufficient)
        grid_barrier(bar, (unsigned)(t + 1) * NB);

        // -------- phase 2: y_t = h_new @ W_lin + b_lin -----------------------
        for (int tile = gwave; tile < (BB / 16) * (VV / 16); tile += NWAVES) {
            const int mt = tile >> 4;        // 0..7
            const int nt = tile & 15;        // 0..15
            v8f acc = {};
            const __bf16* arow = hn + (mt * 16 + l15) * HH + half * 8;
            const __bf16* bp   = Wlin_p + (nt * 64) * 512 + lane * 16;
            for (int kc = 0; kc < HH / 32; ++kc) {
                v8bf alo = *(const v8bf*)(arow + kc * 32);
                v8bf ahi = *(const v8bf*)(arow + kc * 32 + 16);
                v16bf a;
#pragma unroll
                for (int i = 0; i < 8; ++i) { a[i] = alo[i]; a[8 + i] = ahi[i]; }
                v16bf b = *(const v16bf*)(bp + kc * 512);
                acc = __builtin_amdgcn_wmma_f32_16x16x32_bf16(
                          false, a, false, b, (short)0, acc, false, false);
            }
            const int col = nt * 16 + l15;
            const float bl = b_lin[col];
#pragma unroll
            for (int r = 0; r < 8; ++r) {
                const int row = mt * 16 + half * 8 + r;
                out[(row * SS + t) * VV + col] = acc[r] + bl;
            }
        }
    }
}

// ---------------- host entry -------------------------------------------------
extern "C" void kernel_launch(void* const* d_in, const int* in_sizes, int n_in,
                              void* d_out, int out_size, void* d_ws, size_t ws_size,
                              hipStream_t stream) {
    const int*   x     = (const int*)  d_in[0];   // (B,S)
    const float* h0    = (const float*)d_in[1];   // (B,H)
    const float* W_xh  = (const float*)d_in[2];   // (V,H)
    const float* W_hh  = (const float*)d_in[3];   // (H,H)
    const float* b_h   = (const float*)d_in[4];   // (H,)
    const float* W_lin = (const float*)d_in[5];   // (H,V)
    const float* b_lin = (const float*)d_in[6];   // (V,)

    float* out   = (float*)d_out;                 // (B,S,V) then (B,H)
    float* hlast = out + (size_t)BB * SS * VV;

    // workspace layout (~10.5 MB)
    char* ws = (char*)d_ws;
    unsigned* bar    = (unsigned*)ws;                                       // 256 B
    __bf16*   hbuf   = (__bf16*)(ws + 256);                                 // 1 MB
    __bf16*   whh_p  = (__bf16*)(ws + 256 + 2 * BB * HH * 2);               // 8 MB
    __bf16*   wlin_p = (__bf16*)(ws + 256 + 2 * BB * HH * 2 + HH * HH * 2); // 1 MB

    hipMemsetAsync(bar, 0, 256, stream);

    pack_whh <<<(HH * HH + TPB - 1) / TPB, TPB, 0, stream>>>(W_hh, whh_p);
    pack_wlin<<<(HH * VV + TPB - 1) / TPB, TPB, 0, stream>>>(W_lin, wlin_p);
    cvt_h0   <<<(BB * HH + TPB - 1) / TPB, TPB, 0, stream>>>(h0, hbuf);

    rnn_persistent<<<NB, TPB, 0, stream>>>(x, W_xh, b_h, b_lin,
                                           whh_p, wlin_p, hbuf,
                                           out, hlast, bar);
}
